// RGCN_Multi_45243185496737
// MI455X (gfx1250) — compile-verified, matrix operand order checked by reading, not run on hardware
//
#include <hip/hip_runtime.h>

// Problem constants (from reference)
#define BSZ  2500   // nodes / batch
#define TT   400    // time steps
#define DI   16     // input dim
#define HSZ  32     // hidden dim
#define MT   157    // ceil(2500/16) M tiles
#define MPAD (MT*16)    // 2512
#define KCH  80         // ceil(2500/32) -> padded to 80 K-chunks of 32
#define KPAD (KCH*32)   // 2560

typedef __attribute__((ext_vector_type(16))) __bf16 v16bf;
typedef __attribute__((ext_vector_type(8)))  float  v8f;

// Workspace layout (bytes), all 64B-aligned
#define A_SW_OFF   0
#define A_SW_BYTES (MT*KCH*32*16*2)            // 12,861,440 : swizzled bf16 adj
#define B_SW_OFF   (A_SW_OFF + A_SW_BYTES)
#define B_SW_BYTES (KCH*2*32*16*2)             // 163,840 : swizzled bf16 q_t
#define H_OFF      (B_SW_OFF + B_SW_BYTES)
#define H_BYTES    (BSZ*HSZ*4)
#define C_OFF      (H_OFF + H_BYTES)
#define C_BYTES    (BSZ*HSZ*4)
#define G_OFF      (C_OFF + C_BYTES)
#define G_BYTES    (BSZ*4*HSZ*4)               // fp32 gates [2500][128]

// ---- init LSTM state h=c=0 (ws is poisoned, must re-zero each call) ----
__global__ void k_init(float* __restrict__ h, float* __restrict__ c) {
    int i = blockIdx.x * blockDim.x + threadIdx.x;
    if (i < BSZ * HSZ) { h[i] = 0.f; c[i] = 0.f; }
}

// ---- one-time: fp32 adj -> bf16, swizzled into WMMA 16x32 A-operand layout ----
// A layout (16-bit A, 16x32): lane = m + 16*((kin>>3)&1); elem = (kin&7) + 8*(kin>>4)
__global__ void k_swizzle_adj(const float* __restrict__ adj, __bf16* __restrict__ A_sw) {
    int gid = blockIdx.x * blockDim.x + threadIdx.x;
    if (gid >= MPAD * KPAD) return;
    int row = gid / KPAD, col = gid % KPAD;
    float v = (row < BSZ && col < BSZ) ? adj[(size_t)row * BSZ + col] : 0.f;
    int tile = row >> 4, m = row & 15;
    int chunk = col >> 5, kin = col & 31;
    int lane = m + (((kin >> 3) & 1) << 4);
    int e = (kin & 7) + ((kin >> 4) << 3);
    A_sw[((((size_t)tile * KCH + chunk) * 32 + lane) << 4) + e] = (__bf16)v;
}

// ---- per step: gates (fp32) + q_t (tanh, bf16 swizzled B-operand layout) ----
__global__ void k_gates_q(const float* __restrict__ x, const float* __restrict__ W_ih,
                          const float* __restrict__ W_hh, const float* __restrict__ b,
                          const float* __restrict__ W_q, const float* __restrict__ b_q,
                          const float* __restrict__ h, float* __restrict__ gates,
                          __bf16* __restrict__ B_sw, int t) {
    int gid = blockIdx.x * blockDim.x + threadIdx.x;   // KPAD*HSZ threads
    int node = gid >> 5, j = gid & 31;
    if (node >= KPAD) return;
    // B layout mirrors A with N in place of M (half = j/16 selects the N-tile)
    int chunk = node >> 5, kin = node & 31, half = j >> 4, nl = j & 15;
    int lane = nl + (((kin >> 3) & 1) << 4);
    int e = (kin & 7) + ((kin >> 4) << 3);
    size_t slot = ((((size_t)chunk * 2 + half) * 32 + lane) << 4) + e;
    if (node >= BSZ) { B_sw[slot] = (__bf16)0.0f; return; }

    const float* hr = h + (size_t)node * HSZ;
    float q = b_q[j];
    #pragma unroll
    for (int k = 0; k < HSZ; ++k) q += hr[k] * W_q[k * HSZ + j];
    B_sw[slot] = (__bf16)tanhf(q);

    const float* xt = x + (size_t)node * TT * DI + (size_t)t * DI;
    #pragma unroll
    for (int g = 0; g < 4; ++g) {
        int col = j + g * HSZ;
        float a = b[col];
        #pragma unroll
        for (int k = 0; k < DI; ++k)  a += xt[k] * W_ih[k * 4 * HSZ + col];
        #pragma unroll
        for (int k = 0; k < HSZ; ++k) a += hr[k] * W_hh[k * 4 * HSZ + col];
        gates[(size_t)node * 4 * HSZ + col] = a;
    }
}

// ---- per step: agg = adj@q via bf16 WMMA (K split over 4 waves) + fused cell update ----
__global__ void __launch_bounds__(128) k_step(
    const __bf16* __restrict__ A_sw, const __bf16* __restrict__ B_sw,
    const float* __restrict__ gates, float* __restrict__ hbuf, float* __restrict__ cbuf,
    const float* __restrict__ W_d, const float* __restrict__ b_d,
    float* __restrict__ out, int t) {
    __shared__ float red[4][2][8][32];
    int tile = blockIdx.x;
    int wv = threadIdx.x >> 5, lane = threadIdx.x & 31;

    v8f acc0 = {}; v8f acc1 = {};
    const __bf16* Abase = A_sw + (size_t)tile * KCH * 512;   // 512 bf16 per (tile,chunk)
    for (int chunk = wv; chunk < KCH; chunk += 4) {
        v16bf a  = *(const v16bf*)(Abase + (size_t)chunk * 512 + lane * 16);
        v16bf b0 = *(const v16bf*)(B_sw + ((size_t)chunk * 2 + 0) * 512 + lane * 16);
        v16bf b1 = *(const v16bf*)(B_sw + ((size_t)chunk * 2 + 1) * 512 + lane * 16);
        acc0 = __builtin_amdgcn_wmma_f32_16x16x32_bf16(false, a, false, b0,
                                                       (short)0, acc0, false, false);
        acc1 = __builtin_amdgcn_wmma_f32_16x16x32_bf16(false, a, false, b1,
                                                       (short)0, acc1, false, false);
    }
    #pragma unroll
    for (int r = 0; r < 8; ++r) {
        red[wv][0][r][lane] = acc0[r];
        red[wv][1][r][lane] = acc1[r];
    }
    __syncthreads();
    if (wv != 0) return;

    // C/D layout: lane l, vgpr r -> M = r + 8*(l>=16), N = l&15 (acc1: N += 16)
    float wd0 = W_d[lane & 15];
    float wd1 = W_d[(lane & 15) + 16];
    float outacc[8];
    #pragma unroll
    for (int r = 0; r < 8; ++r) outacc[r] = 0.f;
    int mhi = (lane >> 4) << 3;
    int n0 = lane & 15;

    #pragma unroll
    for (int r = 0; r < 8; ++r) {
        int row = tile * 16 + r + mhi;
        if (row >= BSZ) continue;
        #pragma unroll
        for (int half = 0; half < 2; ++half) {
            float agg = red[0][half][r][lane] + red[1][half][r][lane]
                      + red[2][half][r][lane] + red[3][half][r][lane];
            int n = n0 + 16 * half;
            const float* g = gates + (size_t)row * 128;
            float it = 1.f / (1.f + __expf(-g[n]));
            float ft = 1.f / (1.f + __expf(-g[n + 32]));
            float gt = tanhf(g[n + 64]);
            float ot = 1.f / (1.f + __expf(-g[n + 96]));
            float cold = cbuf[(size_t)row * HSZ + n];
            float cn = ft * (cold + agg) + it * gt;
            float hn = ot * tanhf(cn);
            cbuf[(size_t)row * HSZ + n] = cn;
            hbuf[(size_t)row * HSZ + n] = hn;
            outacc[r] += hn * (half ? wd1 : wd0);
        }
    }
    // reduce h.W_d across the 16 lanes of each half-wave (rows differ per group)
    #pragma unroll
    for (int r = 0; r < 8; ++r) {
        outacc[r] += __shfl_xor(outacc[r], 1, 32);
        outacc[r] += __shfl_xor(outacc[r], 2, 32);
        outacc[r] += __shfl_xor(outacc[r], 4, 32);
        outacc[r] += __shfl_xor(outacc[r], 8, 32);
    }
    if ((lane & 15) == 0) {
        float bd = b_d[0];
        #pragma unroll
        for (int r = 0; r < 8; ++r) {
            int row = tile * 16 + r + mhi;
            if (row < BSZ) out[(size_t)row * TT + t] = outacc[r] + bd;
        }
    }
}

extern "C" void kernel_launch(void* const* d_in, const int* in_sizes, int n_in,
                              void* d_out, int out_size, void* d_ws, size_t ws_size,
                              hipStream_t stream) {
    (void)in_sizes; (void)n_in; (void)out_size; (void)ws_size;
    const float* x    = (const float*)d_in[0];
    const float* adj  = (const float*)d_in[1];
    const float* W_ih = (const float*)d_in[2];
    const float* W_hh = (const float*)d_in[3];
    const float* b    = (const float*)d_in[4];
    const float* W_q  = (const float*)d_in[5];
    const float* b_q  = (const float*)d_in[6];
    const float* W_d  = (const float*)d_in[7];
    const float* b_d  = (const float*)d_in[8];
    float* out = (float*)d_out;

    char* ws = (char*)d_ws;
    __bf16* A_sw  = (__bf16*)(ws + A_SW_OFF);
    __bf16* B_sw  = (__bf16*)(ws + B_SW_OFF);
    float*  hbuf  = (float*)(ws + H_OFF);
    float*  cbuf  = (float*)(ws + C_OFF);
    float*  gates = (float*)(ws + G_OFF);

    hipLaunchKernelGGL(k_init, dim3((BSZ * HSZ + 255) / 256), dim3(256), 0, stream,
                       hbuf, cbuf);
    hipLaunchKernelGGL(k_swizzle_adj, dim3((MPAD * KPAD) / 256), dim3(256), 0, stream,
                       adj, A_sw);
    for (int t = 0; t < TT; ++t) {
        hipLaunchKernelGGL(k_gates_q, dim3((KPAD * HSZ) / 256), dim3(256), 0, stream,
                           x, W_ih, W_hh, b, W_q, b_q, hbuf, gates, B_sw, t);
        hipLaunchKernelGGL(k_step, dim3(MT), dim3(128), 0, stream,
                           A_sw, B_sw, gates, hbuf, cbuf, W_d, b_d, out, t);
    }
}
